// BaseModel_31224412242783
// MI455X (gfx1250) — compile-verified, matrix operand order checked by reading, not run on hardware
//
#include <hip/hip_runtime.h>
#include <math.h>

#define B64 64
#define TT  700
#define RROWS (B64*TT)   // 44800 rows, multiple of 16

typedef __attribute__((ext_vector_type(16))) __bf16          v16bf;
typedef __attribute__((ext_vector_type(8)))  float           v8f;
typedef __attribute__((ext_vector_type(8)))  unsigned short  v8u;
typedef __attribute__((ext_vector_type(16))) unsigned short  v16u;

__device__ __forceinline__ unsigned short f2bf(float f) {
  unsigned u = __float_as_uint(f);
  u += 0x7FFFu + ((u >> 16) & 1u);          // round-to-nearest-even
  return (unsigned short)(u >> 16);
}
__device__ __forceinline__ float bf2f(unsigned short h) {
  return __uint_as_float(((unsigned)h) << 16);
}
__device__ __forceinline__ float sigm(float x) { return 1.0f / (1.0f + __expf(-x)); }

// ---------------------------------------------------------------------------
// Embedding replace: xmod[b,c,t] = (c<22) ? emb[argmax_c' x[b,c',t], c] : x[b,c,t]
// ---------------------------------------------------------------------------
__global__ void k_embed(const float* __restrict__ x, const float* __restrict__ emb,
                        float* __restrict__ xmod) {
  int gid = blockIdx.x * blockDim.x + threadIdx.x;
  if (gid >= RROWS) return;
  int b = gid / TT, t = gid % TT;
  const float* xb = x + ((size_t)b * 51) * TT + t;
  int best = 0; float bv = xb[0];
  for (int c = 1; c < 22; ++c) { float v = xb[(size_t)c * TT]; if (v > bv) { bv = v; best = c; } }
  float* xo = xmod + ((size_t)b * 51) * TT + t;
  for (int c = 0;  c < 22; ++c) xo[(size_t)c * TT] = emb[best * 22 + c];
  for (int c = 22; c < 51; ++c) xo[(size_t)c * TT] = xb[(size_t)c * TT];
}

// ---------------------------------------------------------------------------
// Conv1d(3)+Conv1d(5)+passthrough, relu; bf16 rows [t*B+b, 256] and C1[:,0:251]
// ---------------------------------------------------------------------------
__global__ void k_conv(const float* __restrict__ xmod,
                       const float* __restrict__ w3, const float* __restrict__ b3,
                       const float* __restrict__ w5, const float* __restrict__ b5,
                       unsigned short* __restrict__ X0, unsigned short* __restrict__ C1) {
  int gid = blockIdx.x * blockDim.x + threadIdx.x;
  if (gid >= RROWS) return;
  int b = gid / TT, t = gid % TT;
  const float* xb = xmod + ((size_t)b * 51) * TT;
  size_t row = (size_t)t * B64 + b;
  unsigned short* x0 = X0 + row * 256;
  unsigned short* c1 = C1 + row * 768;
  for (int c = 0; c < 51; ++c) {
    float v = fmaxf(xb[(size_t)c * TT + t], 0.f);
    unsigned short h = f2bf(v); x0[c] = h; c1[c] = h;
  }
  for (int o = 0; o < 100; ++o) {               // K=3, pad 1
    float acc = b3[o];
    for (int k = 0; k < 3; ++k) {
      int tt = t + k - 1; if (tt < 0 || tt >= TT) continue;
      const float* wk = w3 + ((size_t)o * 51) * 3 + k;
      for (int c = 0; c < 51; ++c) acc += xb[(size_t)c * TT + tt] * wk[(size_t)c * 3];
    }
    unsigned short h = f2bf(fmaxf(acc, 0.f)); x0[51 + o] = h; c1[51 + o] = h;
  }
  for (int o = 0; o < 100; ++o) {               // K=5, pad 2
    float acc = b5[o];
    for (int k = 0; k < 5; ++k) {
      int tt = t + k - 2; if (tt < 0 || tt >= TT) continue;
      const float* wk = w5 + ((size_t)o * 51) * 5 + k;
      for (int c = 0; c < 51; ++c) acc += xb[(size_t)c * TT + tt] * wk[(size_t)c * 5];
    }
    unsigned short h = f2bf(fmaxf(acc, 0.f)); x0[151 + o] = h; c1[151 + o] = h;
  }
  for (int c = 251; c < 256; ++c) x0[c] = 0;
  for (int c = 751; c < 768; ++c) c1[c] = 0;
}

// ---------------------------------------------------------------------------
// Weight/bias padding into gate-grouped, zero-padded bf16 tiles
// dst row rd -> (g = rd/Hd, j = rd%Hd); src row g*Hs+j if j<Hs else zeros
// ---------------------------------------------------------------------------
__global__ void k_padw(const float* __restrict__ src, unsigned short* __restrict__ dst,
                       int groups, int Hs, int Hd, int Cs, int Cd) {
  int total = groups * Hd * Cd;
  int gid = blockIdx.x * blockDim.x + threadIdx.x;
  if (gid >= total) return;
  int c = gid % Cd; int rd = gid / Cd; int g = rd / Hd; int j = rd % Hd;
  float v = 0.f;
  if (j < Hs && c < Cs) v = src[((size_t)(g * Hs + j)) * Cs + c];
  dst[gid] = f2bf(v);
}

__global__ void k_padb(const float* __restrict__ src, float* __restrict__ dst,
                       int groups, int Hs, int Hd) {
  int total = groups * Hd;
  int gid = blockIdx.x * blockDim.x + threadIdx.x;
  if (gid >= total) return;
  int g = gid / Hd, j = gid % Hd;
  dst[gid] = (j < Hs) ? src[g * Hs + j] : 0.f;
}

// ---------------------------------------------------------------------------
// WMMA bf16 GEMM:  C[R,N] = act(A[R,K] * W[N,K]^T + bias)
// one 16x16 output tile per wave; K multiple of 32; fp32 and/or bf16 outputs
// ---------------------------------------------------------------------------
__global__ void k_gemm(const unsigned short* __restrict__ A, int lda,
                       const unsigned short* __restrict__ W, int ldw,
                       const float* __restrict__ bias,
                       float* __restrict__ Cf, unsigned short* __restrict__ Cb, int ldc,
                       int ntiles, int tiles, int K, int relu) {
  int wid = blockIdx.x * (blockDim.x >> 5) + (threadIdx.x >> 5);
  if (wid >= tiles) return;
  int lane = threadIdx.x & 31;
  int mt = wid / ntiles, nt = wid % ntiles;
  int m = lane & 15, kh = lane >> 4;
  const unsigned short* arow = A + ((size_t)(mt * 16 + m)) * lda;
  const unsigned short* wrow = W + ((size_t)(nt * 16 + m)) * ldw;
  v8f acc = {0.f, 0.f, 0.f, 0.f, 0.f, 0.f, 0.f, 0.f};
  for (int k = 0; k < K; k += 32) {
    v8u a0 = *(const v8u*)(arow + k + 8 * kh);
    v8u a1 = *(const v8u*)(arow + k + 16 + 8 * kh);
    v8u b0 = *(const v8u*)(wrow + k + 8 * kh);
    v8u b1 = *(const v8u*)(wrow + k + 16 + 8 * kh);
    v16u au, bu;
#pragma unroll
    for (int i = 0; i < 8; ++i) { au[i] = a0[i]; au[i + 8] = a1[i]; bu[i] = b0[i]; bu[i + 8] = b1[i]; }
    acc = __builtin_amdgcn_wmma_f32_16x16x32_bf16(false, __builtin_bit_cast(v16bf, au),
                                                  false, __builtin_bit_cast(v16bf, bu),
                                                  (short)0, acc, false, false);
  }
  int n = lane & 15;
#pragma unroll
  for (int v = 0; v < 8; ++v) {
    int row = mt * 16 + kh * 8 + v;
    int col = nt * 16 + n;
    float val = acc[v] + bias[col];
    if (relu) val = fmaxf(val, 0.f);
    if (Cf) Cf[(size_t)row * ldc + col] = val;
    if (Cb) Cb[(size_t)row * ldc + col] = f2bf(val);
  }
}

// ---------------------------------------------------------------------------
// Persistent GRU recurrence, one workgroup (32 waves), state in registers,
// bf16 state copy in LDS feeds v_wmma each step. xg[T*B, 3*Hp] gate-padded.
// rev: read xg at row T-1-t, write hs at row t (loop order, like reference).
// ---------------------------------------------------------------------------
__global__ __launch_bounds__(1024)
void k_gru(const float* __restrict__ xg, const unsigned short* __restrict__ whh,
           const float* __restrict__ bhh, float* __restrict__ hs, int Hp, int rev) {
  __shared__ unsigned short hb[B64 * 512];      // bf16 hidden state (64KB)
  const int tid = threadIdx.x;
  const int G3 = 3 * Hp;
  for (int i = tid; i < B64 * Hp; i += 1024) hb[i] = 0;
  float hreg[4][8];
#pragma unroll
  for (int a = 0; a < 4; ++a)
#pragma unroll
    for (int v = 0; v < 8; ++v) hreg[a][v] = 0.f;
  __syncthreads();

  const int lane = tid & 31, wave = tid >> 5;
  const int m = lane & 15, kh = lane >> 4, nn = lane & 15;
  const int JB = Hp >> 4;
  const int units = (B64 / 16) * JB;            // 64 (Hp=256) or 128 (Hp=512)

  for (int t = 0; t < TT; ++t) {
    int trow = rev ? (TT - 1 - t) : t;
    const float* xgt = xg + (size_t)trow * B64 * G3;
    int ui = 0;
    for (int u = wave; u < units; u += 32, ++ui) {
      int mt = u / JB, jb = u % JB;
      int b0 = mt * 16, j0 = jb * 16;
      v8f ar = {0.f,0.f,0.f,0.f,0.f,0.f,0.f,0.f};
      v8f az = ar, an = ar;
      const unsigned short* hrow = hb + (size_t)(b0 + m) * Hp;
      const unsigned short* wr = whh + (size_t)(0 * Hp + j0 + m) * Hp;
      const unsigned short* wz = whh + (size_t)(1 * Hp + j0 + m) * Hp;
      const unsigned short* wn = whh + (size_t)(2 * Hp + j0 + m) * Hp;
      for (int k = 0; k < Hp; k += 32) {
        v8u a0 = *(const v8u*)(hrow + k + 8 * kh);
        v8u a1 = *(const v8u*)(hrow + k + 16 + 8 * kh);
        v16u au;
#pragma unroll
        for (int i = 0; i < 8; ++i) { au[i] = a0[i]; au[i + 8] = a1[i]; }
        v16bf av = __builtin_bit_cast(v16bf, au);
        {
          v8u c0 = *(const v8u*)(wr + k + 8 * kh), c1 = *(const v8u*)(wr + k + 16 + 8 * kh);
          v16u bu;
#pragma unroll
          for (int i = 0; i < 8; ++i) { bu[i] = c0[i]; bu[i + 8] = c1[i]; }
          ar = __builtin_amdgcn_wmma_f32_16x16x32_bf16(false, av, false,
               __builtin_bit_cast(v16bf, bu), (short)0, ar, false, false);
        }
        {
          v8u c0 = *(const v8u*)(wz + k + 8 * kh), c1 = *(const v8u*)(wz + k + 16 + 8 * kh);
          v16u bu;
#pragma unroll
          for (int i = 0; i < 8; ++i) { bu[i] = c0[i]; bu[i + 8] = c1[i]; }
          az = __builtin_amdgcn_wmma_f32_16x16x32_bf16(false, av, false,
               __builtin_bit_cast(v16bf, bu), (short)0, az, false, false);
        }
        {
          v8u c0 = *(const v8u*)(wn + k + 8 * kh), c1 = *(const v8u*)(wn + k + 16 + 8 * kh);
          v16u bu;
#pragma unroll
          for (int i = 0; i < 8; ++i) { bu[i] = c0[i]; bu[i + 8] = c1[i]; }
          an = __builtin_amdgcn_wmma_f32_16x16x32_bf16(false, av, false,
               __builtin_bit_cast(v16bf, bu), (short)0, an, false, false);
        }
      }
#pragma unroll
      for (int v = 0; v < 8; ++v) {
        int b = b0 + kh * 8 + v;
        int j = j0 + nn;
        const float* xr = xgt + (size_t)b * G3;
        float gr  = xr[j]          + ar[v] + bhh[j];
        float gz  = xr[Hp + j]     + az[v] + bhh[Hp + j];
        float hgn =                  an[v] + bhh[2 * Hp + j];
        float r = sigm(gr);
        float z = sigm(gz);
        float nv = tanhf(xr[2 * Hp + j] + r * hgn);
        hreg[ui][v] = (1.f - z) * nv + z * hreg[ui][v];
      }
    }
    __syncthreads();                             // all reads of hb done
    ui = 0;
    for (int u = wave; u < units; u += 32, ++ui) {
      int mt = u / JB, jb = u % JB;
      int b0 = mt * 16, j0 = jb * 16;
#pragma unroll
      for (int v = 0; v < 8; ++v) {
        int b = b0 + kh * 8 + v;
        int j = j0 + nn;
        float val = hreg[ui][v];
        hb[(size_t)b * Hp + j] = f2bf(val);
        hs[((size_t)t * B64 + b) * Hp + j] = val;
      }
    }
    __syncthreads();                             // hb updated for next step
  }
}

// ---------------------------------------------------------------------------
// Concat / sum combiners
// ---------------------------------------------------------------------------
__global__ void k_comb1(const float* __restrict__ F, const float* __restrict__ Bw,
                        unsigned short* __restrict__ C1, unsigned short* __restrict__ C2) {
  int gid = blockIdx.x * blockDim.x + threadIdx.x;
  if (gid >= RROWS * 250) return;
  int j = gid % 250; size_t row = gid / 250;
  unsigned short f = f2bf(F[row * 256 + j]);
  unsigned short b = f2bf(Bw[row * 256 + j]);
  C1[row * 768 + 251 + j] = f;
  C1[row * 768 + 501 + j] = b;
  C2[row * 1024 + j]       = f;
  C2[row * 1024 + 250 + j] = b;
}

__global__ void k_comb2(const float* __restrict__ F, const float* __restrict__ Bw,
                        unsigned short* __restrict__ C2) {
  int gid = blockIdx.x * blockDim.x + threadIdx.x;
  if (gid >= RROWS * 524) return;
  int j = gid % 524; size_t row = gid / 524;
  C2[row * 1024 + 500 + j] = (j < 500) ? f2bf(F[row * 512 + j] + Bw[row * 512 + j]) : 0;
}

__global__ void k_comb3(const float* __restrict__ F, const float* __restrict__ Bw,
                        unsigned short* __restrict__ S) {
  int gid = blockIdx.x * blockDim.x + threadIdx.x;
  if (gid >= RROWS * 512) return;
  int j = gid % 512; size_t row = gid / 512;
  S[row * 512 + j] = (j < 500) ? f2bf(F[row * 512 + j] + Bw[row * 512 + j]) : 0;
}

// ---------------------------------------------------------------------------
// fc2 (N=9) + layout remap [t*B+b,128] -> out[b,T,9]
// ---------------------------------------------------------------------------
__global__ void k_fc2(const unsigned short* __restrict__ Y1, const float* __restrict__ w2,
                      const float* __restrict__ b2, float* __restrict__ out) {
  int gid = blockIdx.x * blockDim.x + threadIdx.x;
  if (gid >= RROWS * 9) return;
  int o = gid % 9; size_t row = gid / 9;
  int t = (int)(row / B64), b = (int)(row % B64);
  float acc = b2[o];
  const unsigned short* y = Y1 + row * 128;
  const float* w = w2 + o * 128;
  for (int k = 0; k < 128; ++k) acc += bf2f(y[k]) * w[k];
  out[((size_t)b * TT + t) * 9 + o] = acc;
}

// ---------------------------------------------------------------------------
extern "C" void kernel_launch(void* const* d_in, const int* in_sizes, int n_in,
                              void* d_out, int out_size, void* d_ws, size_t ws_size,
                              hipStream_t stream) {
  (void)in_sizes; (void)n_in; (void)out_size; (void)ws_size;
  const float* x    = (const float*)d_in[0];
  const float* emb  = (const float*)d_in[1];
  const float* w3   = (const float*)d_in[2];
  const float* b3   = (const float*)d_in[3];
  const float* w5   = (const float*)d_in[4];
  const float* b5   = (const float*)d_in[5];
  const float* w11  = (const float*)d_in[6];
  const float* b11  = (const float*)d_in[7];
  const float* w12  = (const float*)d_in[8];
  const float* b12  = (const float*)d_in[9];
  const float *g_wih[6], *g_whh[6], *g_bih[6], *g_bhh[6];
  for (int i = 0; i < 6; ++i) {
    g_wih[i] = (const float*)d_in[10 + 4 * i + 0];
    g_whh[i] = (const float*)d_in[10 + 4 * i + 1];
    g_bih[i] = (const float*)d_in[10 + 4 * i + 2];
    g_bhh[i] = (const float*)d_in[10 + 4 * i + 3];
  }
  const float* fc1w = (const float*)d_in[34];
  const float* fc1b = (const float*)d_in[35];
  const float* fc2w = (const float*)d_in[36];
  const float* fc2b = (const float*)d_in[37];
  float* out = (float*)d_out;

  const size_t R = RROWS;
  char* ws = (char*)d_ws;
  size_t off = 0;
  auto alloc = [&](size_t bytes) -> char* {
    char* p = ws + off; off += (bytes + 255) & ~(size_t)255; return p;
  };

  float*          xmod = (float*)alloc((size_t)B64 * 51 * TT * 4);
  unsigned short* X0   = (unsigned short*)alloc(R * 256 * 2);
  unsigned short* C1   = (unsigned short*)alloc(R * 768 * 2);
  unsigned short* C2   = (unsigned short*)alloc(R * 1024 * 2);
  unsigned short* X1   = (unsigned short*)alloc(R * 512 * 2);   // also reused for X2
  unsigned short* S    = (unsigned short*)alloc(R * 512 * 2);
  unsigned short* Y1   = (unsigned short*)alloc(R * 128 * 2);
  float*          xgF  = (float*)alloc(R * 1536 * 4);
  float*          xgB  = (float*)alloc(R * 1536 * 4);
  float*          Fh   = (float*)alloc(R * 512 * 4);
  float*          Bh   = (float*)alloc(R * 512 * 4);

  unsigned short *Wih[6], *Whh[6]; float *Bih[6], *Bhh[6];
  for (int i = 0; i < 6; ++i) {
    size_t wsz = (i < 2) ? (size_t)768 * 256 : (size_t)1536 * 512;
    size_t bsz = (i < 2) ? 768 : 1536;
    Wih[i] = (unsigned short*)alloc(wsz * 2);
    Whh[i] = (unsigned short*)alloc(wsz * 2);
    Bih[i] = (float*)alloc(bsz * 4);
    Bhh[i] = (float*)alloc(bsz * 4);
  }
  unsigned short* W11p = (unsigned short*)alloc((size_t)512 * 768 * 2);
  unsigned short* W12p = (unsigned short*)alloc((size_t)512 * 1024 * 2);
  unsigned short* FC1p = (unsigned short*)alloc((size_t)128 * 512 * 2);
  float* B11p  = (float*)alloc(512 * 4);
  float* B12p  = (float*)alloc(512 * 4);
  float* FC1bp = (float*)alloc(128 * 4);

  auto cdiv = [](long a, long b) { return (int)((a + b - 1) / b); };

  // ---- weight prep ----
  for (int i = 0; i < 6; ++i) {
    int Hs = (i < 2) ? 250 : 500, Hd = (i < 2) ? 256 : 512;
    int CsW = (i < 2) ? 251 : 500, CsH = (i < 2) ? 250 : 500, Cd = (i < 2) ? 256 : 512;
    k_padw<<<cdiv((long)3 * Hd * Cd, 256), 256, 0, stream>>>(g_wih[i], Wih[i], 3, Hs, Hd, CsW, Cd);
    k_padw<<<cdiv((long)3 * Hd * Cd, 256), 256, 0, stream>>>(g_whh[i], Whh[i], 3, Hs, Hd, CsH, Cd);
    k_padb<<<cdiv((long)3 * Hd, 256), 256, 0, stream>>>(g_bih[i], Bih[i], 3, Hs, Hd);
    k_padb<<<cdiv((long)3 * Hd, 256), 256, 0, stream>>>(g_bhh[i], Bhh[i], 3, Hs, Hd);
  }
  k_padw<<<cdiv((long)512 * 768, 256), 256, 0, stream>>>(w11, W11p, 1, 500, 512, 751, 768);
  k_padw<<<cdiv((long)512 * 1024, 256), 256, 0, stream>>>(w12, W12p, 1, 500, 512, 1000, 1024);
  k_padw<<<cdiv((long)128 * 512, 256), 256, 0, stream>>>(fc1w, FC1p, 1, 128, 128, 500, 512);
  k_padb<<<cdiv(512, 256), 256, 0, stream>>>(b11, B11p, 1, 500, 512);
  k_padb<<<cdiv(512, 256), 256, 0, stream>>>(b12, B12p, 1, 500, 512);
  k_padb<<<cdiv(128, 256), 256, 0, stream>>>(fc1b, FC1bp, 1, 128, 128);

  auto gemm = [&](const unsigned short* A, int lda, const unsigned short* W, int ldw,
                  const float* bias, float* Cf, unsigned short* Cb, int N, int K, int relu) {
    long tiles = (long)(R / 16) * (N / 16);
    k_gemm<<<cdiv(tiles, 8), 256, 0, stream>>>(A, lda, W, ldw, bias, Cf, Cb, N, N / 16,
                                               (int)tiles, K, relu);
  };

  // ---- front-end ----
  k_embed<<<cdiv((long)R, 256), 256, 0, stream>>>(x, emb, xmod);
  k_conv<<<cdiv((long)R, 256), 256, 0, stream>>>(xmod, w3, b3, w5, b5, X0, C1);

  // ---- GRU layer 1 (H=250, Hp=256) ----
  gemm(X0, 256, Wih[0], 256, Bih[0], xgF, nullptr, 768, 256, 0);
  gemm(X0, 256, Wih[1], 256, Bih[1], xgB, nullptr, 768, 256, 0);
  k_gru<<<1, 1024, 0, stream>>>(xgF, Whh[0], Bhh[0], Fh, 256, 0);
  k_gru<<<1, 1024, 0, stream>>>(xgB, Whh[1], Bhh[1], Bh, 256, 1);
  k_comb1<<<cdiv((long)R * 250, 256), 256, 0, stream>>>(Fh, Bh, C1, C2);

  // ---- 1x1 mix w11 + GRU layer 2 (H=500, Hp=512) ----
  gemm(C1, 768, W11p, 768, B11p, nullptr, X1, 512, 768, 1);
  gemm(X1, 512, Wih[2], 512, Bih[2], xgF, nullptr, 1536, 512, 0);
  gemm(X1, 512, Wih[3], 512, Bih[3], xgB, nullptr, 1536, 512, 0);
  k_gru<<<1, 1024, 0, stream>>>(xgF, Whh[2], Bhh[2], Fh, 512, 0);
  k_gru<<<1, 1024, 0, stream>>>(xgB, Whh[3], Bhh[3], Bh, 512, 1);
  k_comb2<<<cdiv((long)R * 524, 256), 256, 0, stream>>>(Fh, Bh, C2);

  // ---- 1x1 mix w12 + GRU layer 3 ----
  gemm(C2, 1024, W12p, 1024, B12p, nullptr, X1, 512, 1024, 1);  // X2 reuses X1
  gemm(X1, 512, Wih[4], 512, Bih[4], xgF, nullptr, 1536, 512, 0);
  gemm(X1, 512, Wih[5], 512, Bih[5], xgB, nullptr, 1536, 512, 0);
  k_gru<<<1, 1024, 0, stream>>>(xgF, Whh[4], Bhh[4], Fh, 512, 0);
  k_gru<<<1, 1024, 0, stream>>>(xgB, Whh[5], Bhh[5], Bh, 512, 1);
  k_comb3<<<cdiv((long)R * 512, 256), 256, 0, stream>>>(Fh, Bh, S);

  // ---- head ----
  gemm(S, 512, FC1p, 512, FC1bp, nullptr, Y1, 128, 512, 1);
  k_fc2<<<cdiv((long)R * 9, 256), 256, 0, stream>>>(Y1, fc2w, fc2b, out);
}
